// SA_ABMILP_84112639525171
// MI455X (gfx1250) — matrix operand order repaired and checked by reading, not compiled, not run on hardware
//
#include <hip/hip_runtime.h>
#include <hip/hip_bf16.h>

typedef __attribute__((ext_vector_type(16))) _Float16 v16h;
typedef __attribute__((ext_vector_type(8)))  float    v8f;
typedef __attribute__((ext_vector_type(2)))  float    v2f;
typedef __attribute__((ext_vector_type(4)))  int      v4i;

#define NROWS 8192
#define DIN   1024
#define F1    256
#define F2    128
#define F3    64
#define DQK   8

// ---------------- async global->LDS copy (gfx1250), guarded ----------------
#if defined(__has_builtin)
#if __has_builtin(__builtin_amdgcn_global_load_async_to_lds_b128)
#define HAVE_ASYNC_LDS 1
#endif
#endif

typedef __attribute__((address_space(1))) v4i gv4i;
typedef __attribute__((address_space(3))) v4i lv4i;

static __device__ __forceinline__ void copy16_g2l(const void* g, void* l) {
#ifdef HAVE_ASYNC_LDS
  __builtin_amdgcn_global_load_async_to_lds_b128((gv4i*)(void*)g, (lv4i*)l, 0, 0);
#else
  *(uint4*)l = *(const uint4*)g;
#endif
}
static __device__ __forceinline__ void copy_wait() {
#ifdef HAVE_ASYNC_LDS
#if __has_builtin(__builtin_amdgcn_s_wait_asynccnt)
  __builtin_amdgcn_s_wait_asynccnt(0);
#endif
#endif
}

// ---------------- WMMA wrappers ----------------
static __device__ __forceinline__ v8f wmma_f16(v16h a, v16h b, v8f c) {
  return __builtin_amdgcn_wmma_f32_16x16x32_f16(false, a, false, b, (short)0, c, false, false);
}
static __device__ __forceinline__ v8f wmma_f32k4(v2f a, v2f b, v8f c) {
  return __builtin_amdgcn_wmma_f32_16x16x4_f32(false, a, false, b, (short)0, c, false, false);
}

// ---------------- fragment swizzle (per-lane contiguous, ISA 7.12.2) ----------
// A 16x32 f16 fragment: lane = m + 16*((k>>3)&1); slot = (k&7) + 8*((k>>4)&1)
static __device__ __forceinline__ int a_swz(int m, int k) {
  return (m + (((k >> 3) & 1) << 4)) * 16 + ((k & 7) + (((k >> 4) & 1) << 3));
}
// B 32x16 f16 fragment: lane = n + 16*((k>>3)&1); slot = (k&7) + 8*((k>>4)&1)
static __device__ __forceinline__ int b_swz(int k, int n) {
  return (n + (((k >> 3) & 1) << 4)) * 16 + ((k & 7) + (((k >> 4) & 1) << 3));
}
// fragment load: 32 bytes per lane, two b128 LDS loads
static __device__ __forceinline__ v16h load_frag(const _Float16* frag, int lane) {
  const uint4* p = (const uint4*)(frag + lane * 16);
  union { uint4 u[2]; v16h v; } t;
  t.u[0] = p[0];
  t.u[1] = p[1];
  return t.v;
}

// =====================================================================
// Kernel 0: pre-swizzle all weights into f16 fragment layout (run once/call)
// =====================================================================
__global__ __launch_bounds__(256) void prep_kernel(
    const float* __restrict__ W1, const float* __restrict__ W2,
    const float* __restrict__ W3, const float* __restrict__ Wq,
    const float* __restrict__ Wk, const float* __restrict__ Wv,
    _Float16* __restrict__ W1sw, _Float16* __restrict__ W2sw,
    _Float16* __restrict__ W3sw, _Float16* __restrict__ Wvqksw) {
  const int i = blockIdx.x * 256 + threadIdx.x;
  if (i < DIN * F1) {  // W1: 64-K chunks, 16 col tiles
    int k = i >> 8, n = i & 255;
    int kc = k >> 6, kk = k & 63;
    int idx = (((kc * 2 + (kk >> 5)) * 16 + (n >> 4)) << 9) + b_swz(kk & 31, n & 15);
    W1sw[idx] = (_Float16)W1[i];
  }
  if (i < F1 * F2) {  // W2: 64-K chunks, 8 col tiles
    int k = i >> 7, n = i & 127;
    int kc = k >> 6, kk = k & 63;
    int idx = (((kc * 2 + (kk >> 5)) * 8 + (n >> 4)) << 9) + b_swz(kk & 31, n & 15);
    W2sw[idx] = (_Float16)W2[i];
  }
  if (i < F2 * F3) {  // W3: 4 K-steps, 4 col tiles
    int k = i >> 6, n = i & 63;
    int idx = ((((k >> 5) * 4) + (n >> 4)) << 9) + b_swz(k & 31, n & 15);
    W3sw[idx] = (_Float16)W3[i];
  }
  if (i < F3 * 80) {  // [Wv | Wq | Wk] 64x80: 2 K-steps, 5 col tiles
    int k = i / 80, n = i - (i / 80) * 80;
    float v = (n < 64) ? Wv[k * F3 + n]
                       : ((n < 72) ? Wq[k * DQK + (n - 64)] : Wk[k * DQK + (n - 72)]);
    int idx = ((((k >> 5) * 5) + (n >> 4)) << 9) + b_swz(k & 31, n & 15);
    Wvqksw[idx] = (_Float16)v;
  }
}

// =====================================================================
// Kernel 1: fused MLP + Q/K/V projections. 64 rows/block, 8 waves.
// All LDS matrices live in fragment layout; weight staging is async b128.
// =====================================================================
__global__ __launch_bounds__(256) void mlp_kernel(
    const float* __restrict__ x,
    const _Float16* __restrict__ W1sw, const float* __restrict__ b1,
    const _Float16* __restrict__ W2sw, const float* __restrict__ b2,
    const _Float16* __restrict__ W3sw, const float* __restrict__ b3,
    const _Float16* __restrict__ Wvqksw,
    const float* __restrict__ bq, const float* __restrict__ bk,
    const float* __restrict__ bv,
    float* __restrict__ Hout, float* __restrict__ Qout,
    float* __restrict__ Kout, _Float16* __restrict__ Vsw) {
  __shared__ __align__(16) char smem[49152];
  _Float16* bufA = (_Float16*)smem;            // 32 KB: 32 fragments
  _Float16* bufB = (_Float16*)(smem + 32768);  // 16 KB: 16 fragments

  const int tid = threadIdx.x, lane = tid & 31, wave = tid >> 5;
  const int rg = wave & 3, ch = wave >> 2;
  const int row0 = blockIdx.x * 64;
  const int n = lane & 15, hh = (lane >> 4) & 1;

  // ---------- layer 1: H1[64,256] = relu(x @ W1) ----------
  v8f acc1[8];
#pragma unroll
  for (int t = 0; t < 8; ++t) acc1[t] = (v8f)(0.0f);

  _Float16* sXf  = bufB;  // 8 frags: (rg, ks)
  _Float16* sW1f = bufA;  // 32 frags: (ks, tile)
  for (int kc = 0; kc < DIN / 64; ++kc) {
    const _Float16* wsrc = W1sw + kc * (32 * 512);
    for (int i = tid; i < 2048; i += 256) copy16_g2l(wsrc + i * 8, sW1f + i * 8);
    const float* xs = x + row0 * DIN + kc * 64;
    for (int i = tid; i < 64 * 64; i += 256) {
      int r = i >> 6, k = i & 63;
      sXf[(((r >> 4) * 2 + (k >> 5)) << 9) + a_swz(r & 15, k & 31)] =
          (_Float16)xs[r * DIN + k];
    }
    copy_wait();
    __syncthreads();
#pragma unroll
    for (int ks = 0; ks < 2; ++ks) {
      v16h a = load_frag(sXf + ((rg * 2 + ks) << 9), lane);
#pragma unroll
      for (int t = 0; t < 8; ++t) {
        v16h b = load_frag(sW1f + ((ks * 16 + ch * 8 + t) << 9), lane);
        acc1[t] = wmma_f16(a, b, acc1[t]);
      }
    }
    __syncthreads();
  }
  // relu+bias -> sH1f (32 A-frags: (rg, kc2)) reusing bufA
  _Float16* sH1f = bufA;
#pragma unroll
  for (int t = 0; t < 8; ++t) {
    int col = ch * 128 + t * 16 + n;
    float bb = b1[col];
#pragma unroll
    for (int r = 0; r < 8; ++r) {
      float v = acc1[t][r] + bb;
      v = v > 0.0f ? v : 0.0f;
      sH1f[((rg * 8 + (col >> 5)) << 9) + a_swz(hh * 8 + r, col & 31)] = (_Float16)v;
    }
  }
  __syncthreads();

  // ---------- layer 2: H2[64,128] = relu(H1 @ W2) ----------
  v8f acc2[4];
#pragma unroll
  for (int t = 0; t < 4; ++t) acc2[t] = (v8f)(0.0f);
  _Float16* sW2f = bufB;  // 16 frags: (ks, tile)
  for (int kc = 0; kc < 4; ++kc) {
    const _Float16* wsrc = W2sw + kc * (16 * 512);
    for (int i = tid; i < 1024; i += 256) copy16_g2l(wsrc + i * 8, sW2f + i * 8);
    copy_wait();
    __syncthreads();
#pragma unroll
    for (int ks = 0; ks < 2; ++ks) {
      v16h a = load_frag(sH1f + ((rg * 8 + kc * 2 + ks) << 9), lane);
#pragma unroll
      for (int t = 0; t < 4; ++t) {
        v16h b = load_frag(sW2f + ((ks * 8 + ch * 4 + t) << 9), lane);
        acc2[t] = wmma_f16(a, b, acc2[t]);
      }
    }
    __syncthreads();
  }
  _Float16* sH2f = bufB;  // 16 A-frags: (rg, kc2)
#pragma unroll
  for (int t = 0; t < 4; ++t) {
    int col = ch * 64 + t * 16 + n;
    float bb = b2[col];
#pragma unroll
    for (int r = 0; r < 8; ++r) {
      float v = acc2[t][r] + bb;
      v = v > 0.0f ? v : 0.0f;
      sH2f[((rg * 4 + (col >> 5)) << 9) + a_swz(hh * 8 + r, col & 31)] = (_Float16)v;
    }
  }
  _Float16* sW3f = bufA;  // 16 frags: (ks, tile), overwrites sH1f
  for (int i = tid; i < 1024; i += 256) copy16_g2l(W3sw + i * 8, sW3f + i * 8);
  copy_wait();
  __syncthreads();

  // ---------- layer 3: H3[64,64] = relu(H2 @ W3) ----------
  v8f acc3[2];
#pragma unroll
  for (int t = 0; t < 2; ++t) acc3[t] = (v8f)(0.0f);
#pragma unroll
  for (int ks = 0; ks < 4; ++ks) {
    v16h a = load_frag(sH2f + ((rg * 4 + ks) << 9), lane);
#pragma unroll
    for (int t = 0; t < 2; ++t) {
      v16h b = load_frag(sW3f + ((ks * 4 + ch * 2 + t) << 9), lane);
      acc3[t] = wmma_f16(a, b, acc3[t]);
    }
  }
  __syncthreads();
  _Float16* sH3f = bufA;  // 8 A-frags: (rg, ks)
#pragma unroll
  for (int t = 0; t < 2; ++t) {
    int col = ch * 32 + t * 16 + n;
    float bb = b3[col];
#pragma unroll
    for (int r = 0; r < 8; ++r) {
      float v = acc3[t][r] + bb;
      v = v > 0.0f ? v : 0.0f;
      sH3f[((rg * 2 + (col >> 5)) << 9) + a_swz(hh * 8 + r, col & 31)] = (_Float16)v;
      Hout[(row0 + rg * 16 + hh * 8 + r) * F3 + col] = v;
    }
  }
  _Float16* sWBf = bufB;  // 10 frags: (ks, tile 0..4)
  for (int i = tid; i < 640; i += 256) copy16_g2l(Wvqksw + i * 8, sWBf + i * 8);
  copy_wait();
  __syncthreads();

  // ---------- Q/K/V: H3 @ [Wv|Wq|Wk] ----------
  const int tstart = (ch == 0) ? 0 : 3;
  const int tcount = (ch == 0) ? 3 : 2;
  v8f accv[3];
#pragma unroll
  for (int t = 0; t < 3; ++t) accv[t] = (v8f)(0.0f);
#pragma unroll
  for (int ks = 0; ks < 2; ++ks) {
    v16h a = load_frag(sH3f + ((rg * 2 + ks) << 9), lane);
    for (int tt = 0; tt < tcount; ++tt) {
      v16h b = load_frag(sWBf + ((ks * 5 + tstart + tt) << 9), lane);
      accv[tt] = wmma_f16(a, b, accv[tt]);
    }
  }
  for (int tt = 0; tt < tcount; ++tt) {
    int col = (tstart + tt) * 16 + n;
#pragma unroll
    for (int r = 0; r < 8; ++r) {
      int row = row0 + rg * 16 + hh * 8 + r;
      float v = accv[tt][r];
      if (col < 64) {
        // V stored pre-swizzled as PV B-fragments per 64-key chunk
        int idx = (((row >> 6) * 8 + ((row >> 5) & 1) * 4 + (col >> 4)) << 9) +
                  b_swz(row & 31, col & 15);
        Vsw[idx] = (_Float16)(v + bv[col]);
      } else if (col < 72) {
        Qout[row * DQK + (col - 64)] = v + bq[col - 64];
      } else {
        Kout[row * DQK + (col - 72)] = v + bk[col - 72];
      }
    }
  }
}

// =====================================================================
// Kernel 2: flash attention. wave = 16 query rows; stream 64-key chunks.
// =====================================================================
__global__ __launch_bounds__(256) void attn_kernel(
    const float* __restrict__ Qg, const float* __restrict__ Kg,
    const _Float16* __restrict__ Vsw, const float* __restrict__ Hin,
    const float* __restrict__ gamma, float* __restrict__ Hnout) {
  __shared__ __align__(16) char smem[19200];
  float*    sK  = (float*)smem;                 // [64][10] f32 (8B-aligned rows)
  _Float16* sVf = (_Float16*)(smem + 2560);     // 8 B-frags (8 KB)
  _Float16* sPf = (_Float16*)(smem + 10752);    // 8 waves x 2 A-frags (16 KB)

  const int tid = threadIdx.x, lane = tid & 31, wave = tid >> 5;
  const int qrow0 = (blockIdx.x * 8 + wave) * 16;
  const int n = lane & 15, hh = (lane >> 4) & 1;
  _Float16* pw = sPf + wave * 1024;

  const int m = lane & 15, kh = (lane >> 4) & 1;
  v2f aQ0 = *(const v2f*)(Qg + (qrow0 + m) * DQK + kh * 2);
  v2f aQ1 = *(const v2f*)(Qg + (qrow0 + m) * DQK + 4 + kh * 2);

  float mrow[8], lrow[8];
#pragma unroll
  for (int r = 0; r < 8; ++r) { mrow[r] = -1e30f; lrow[r] = 0.0f; }
  v8f O[4];
#pragma unroll
  for (int t = 0; t < 4; ++t) O[t] = (v8f)(0.0f);

  for (int kc = 0; kc < NROWS / 64; ++kc) {
    const int key0 = kc * 64;
    for (int i = tid; i < 64 * DQK; i += 256) {
      int r = i >> 3, c = i & 7;
      sK[r * 10 + c] = Kg[(key0 + r) * DQK + c];
    }
    const _Float16* vsrc = Vsw + kc * 4096;
    for (int i = tid; i < 512; i += 256) copy16_g2l(vsrc + i * 8, sVf + i * 8);
    copy_wait();
    __syncthreads();

    // S[16q x 64k] = Q @ K^T (two K=4 steps, f32 WMMA)
    v8f S[4];
#pragma unroll
    for (int t = 0; t < 4; ++t) {
      S[t] = (v8f)(0.0f);
      v2f b0 = *(const v2f*)(sK + (t * 16 + n) * 10 + kh * 2);
      S[t] = wmma_f32k4(aQ0, b0, S[t]);
      v2f b1 = *(const v2f*)(sK + (t * 16 + n) * 10 + 4 + kh * 2);
      S[t] = wmma_f32k4(aQ1, b1, S[t]);
    }
    // online softmax: row r lives across 16 lanes of each half
    float cm[8];
#pragma unroll
    for (int r = 0; r < 8; ++r) {
      float mm = S[0][r];
#pragma unroll
      for (int t = 1; t < 4; ++t) mm = fmaxf(mm, S[t][r]);
#pragma unroll
      for (int off = 8; off >= 1; off >>= 1) mm = fmaxf(mm, __shfl_xor(mm, off, 32));
      cm[r] = mm;
    }
    float sc[8], rs[8];
#pragma unroll
    for (int r = 0; r < 8; ++r) {
      float mn = fmaxf(mrow[r], cm[r]);
      sc[r] = __expf(mrow[r] - mn);
      mrow[r] = mn;
      rs[r] = 0.0f;
    }
#pragma unroll
    for (int t = 0; t < 4; ++t)
#pragma unroll
      for (int r = 0; r < 8; ++r) {
        float p = __expf(S[t][r] - mrow[r]);
        S[t][r] = p;
        rs[r] += p;
      }
#pragma unroll
    for (int r = 0; r < 8; ++r) {
#pragma unroll
      for (int off = 8; off >= 1; off >>= 1) rs[r] += __shfl_xor(rs[r], off, 32);
      lrow[r] = lrow[r] * sc[r] + rs[r];
    }
#pragma unroll
    for (int t = 0; t < 4; ++t)
#pragma unroll
      for (int r = 0; r < 8; ++r) O[t][r] *= sc[r];

    // P -> per-wave LDS tile in A-fragment layout
#pragma unroll
    for (int t = 0; t < 4; ++t)
#pragma unroll
      for (int r = 0; r < 8; ++r) {
        int k = t * 16 + n;
        pw[((k >> 5) << 9) + a_swz(hh * 8 + r, k & 31)] = (_Float16)S[t][r];
      }
#pragma unroll
    for (int ks = 0; ks < 2; ++ks) {
      v16h aP = load_frag(pw + (ks << 9), lane);
#pragma unroll
      for (int t = 0; t < 4; ++t) {
        v16h bV = load_frag(sVf + ((ks * 4 + t) << 9), lane);
        O[t] = wmma_f16(aP, bV, O[t]);
      }
    }
    __syncthreads();
  }

  const float g = gamma[0];
#pragma unroll
  for (int r = 0; r < 8; ++r) {
    float inv = 1.0f / lrow[r];
    int row = qrow0 + hh * 8 + r;
#pragma unroll
    for (int t = 0; t < 4; ++t) {
      int col = t * 16 + n;
      Hnout[row * F3 + col] = g * O[t][r] * inv + Hin[row * F3 + col];
    }
  }
}

// =====================================================================
// Kernel 3: s_i = tanh(Hn_i @ Wa1 + ba1) @ Wa2 + ba2
// =====================================================================
__global__ __launch_bounds__(256) void score_kernel(
    const float* __restrict__ Hn, const float* __restrict__ Wa1,
    const float* __restrict__ ba1, const float* __restrict__ Wa2,
    const float* __restrict__ ba2, float* __restrict__ sArr) {
  __shared__ float sW[F3 * F3];
  __shared__ float sb1[F3];
  __shared__ float sw2[F3];
  const int tid = threadIdx.x;
  for (int i = tid; i < F3 * F3; i += 256) sW[i] = Wa1[i];
  if (tid < F3) { sb1[tid] = ba1[tid]; sw2[tid] = Wa2[tid]; }
  __syncthreads();

  const int row = blockIdx.x * 256 + tid;
  float h[F3];
#pragma unroll
  for (int c = 0; c < F3; ++c) h[c] = Hn[row * F3 + c];
  float s = ba2[0];
  for (int j = 0; j < F3; ++j) {
    float d = sb1[j];
#pragma unroll
    for (int c = 0; c < F3; ++c) d += h[c] * sW[c * F3 + j];
    s += tanhf(d) * sw2[j];
  }
  sArr[row] = s;
}

// =====================================================================
// Kernel 4: softmax pooling + classifier. out[0]=Y_prob, out[1..64]=M
// =====================================================================
__global__ __launch_bounds__(256) void final_kernel(
    const float* __restrict__ Hn, const float* __restrict__ sArr,
    const float* __restrict__ Wc, const float* __restrict__ bc,
    float* __restrict__ out) {
  __shared__ float red[256];
  __shared__ float sM[F3];
  const int tid = threadIdx.x;

  float lmax = -1e30f;
  for (int i = tid; i < NROWS; i += 256) lmax = fmaxf(lmax, sArr[i]);
  red[tid] = lmax;
  __syncthreads();
  for (int s = 128; s > 0; s >>= 1) {
    if (tid < s) red[tid] = fmaxf(red[tid], red[tid + s]);
    __syncthreads();
  }
  const float mx = red[0];
  __syncthreads();

  float lsum = 0.0f;
  for (int i = tid; i < NROWS; i += 256) lsum += __expf(sArr[i] - mx);
  red[tid] = lsum;
  __syncthreads();
  for (int s = 128; s > 0; s >>= 1) {
    if (tid < s) red[tid] += red[tid + s];
    __syncthreads();
  }
  const float invD = 1.0f / red[0];
  if (tid < F3) sM[tid] = 0.0f;
  __syncthreads();

  float pm[F3];
#pragma unroll
  for (int c = 0; c < F3; ++c) pm[c] = 0.0f;
  for (int i = tid; i < NROWS; i += 256) {
    float w = __expf(sArr[i] - mx) * invD;
#pragma unroll
    for (int c = 0; c < F3; ++c) pm[c] += w * Hn[i * F3 + c];
  }
#pragma unroll
  for (int c = 0; c < F3; ++c) atomicAdd(&sM[c], pm[c]);
  __syncthreads();

  if (tid == 0) {
    float y = bc[0];
#pragma unroll
    for (int c = 0; c < F3; ++c) y += sM[c] * Wc[c];
    y = 1.0f / (1.0f + __expf(-y));
    y = fminf(fmaxf(y, 1e-5f), 1.0f - 1e-5f);
    out[0] = y;
  }
  if (tid < F3) out[1 + tid] = sM[tid];
}

// =====================================================================
extern "C" void kernel_launch(void* const* d_in, const int* in_sizes, int n_in,
                              void* d_out, int out_size, void* d_ws, size_t ws_size,
                              hipStream_t stream) {
  (void)in_sizes; (void)n_in; (void)out_size; (void)ws_size;
  const float* x   = (const float*)d_in[0];
  const float* W1  = (const float*)d_in[1];
  const float* b1  = (const float*)d_in[2];
  const float* W2  = (const float*)d_in[3];
  const float* b2  = (const float*)d_in[4];
  const float* W3  = (const float*)d_in[5];
  const float* b3  = (const float*)d_in[6];
  const float* Wq  = (const float*)d_in[7];
  const float* bq  = (const float*)d_in[8];
  const float* Wk  = (const float*)d_in[9];
  const float* bk  = (const float*)d_in[10];
  const float* Wv  = (const float*)d_in[11];
  const float* bv  = (const float*)d_in[12];
  const float* gm  = (const float*)d_in[13];
  const float* Wa1 = (const float*)d_in[14];
  const float* ba1 = (const float*)d_in[15];
  const float* Wa2 = (const float*)d_in[16];
  const float* ba2 = (const float*)d_in[17];
  const float* Wc  = (const float*)d_in[18];
  const float* bc  = (const float*)d_in[19];

  char* ws = (char*)d_ws;
  float*    H      = (float*)(ws + 0);            // 8192*64 f32
  float*    Hn     = (float*)(ws + 2097152);      // 8192*64 f32
  float*    Q      = (float*)(ws + 4194304);      // 8192*8 f32
  float*    Kq     = (float*)(ws + 4456448);      // 8192*8 f32
  _Float16* Vsw    = (_Float16*)(ws + 4718592);   // 8192*64 f16 (swizzled)
  float*    sArr   = (float*)(ws + 5767168);      // 8192 f32
  _Float16* W1sw   = (_Float16*)(ws + 5799936);   // 512 KB
  _Float16* W2sw   = (_Float16*)(ws + 6324224);   // 64 KB
  _Float16* W3sw   = (_Float16*)(ws + 6389760);   // 16 KB
  _Float16* Wvqksw = (_Float16*)(ws + 6406144);   // 10 KB

  prep_kernel<<<(DIN * F1) / 256, 256, 0, stream>>>(W1, W2, W3, Wq, Wk, Wv,
                                                    W1sw, W2sw, W3sw, Wvqksw);
  mlp_kernel<<<NROWS / 64, 256, 0, stream>>>(x, W1sw, b1, W2sw, b2, W3sw, b3,
                                             Wvqksw, bq, bk, bv, H, Q, Kq, Vsw);
  attn_kernel<<<NROWS / 128, 256, 0, stream>>>(Q, Kq, Vsw, H, gm, Hn);
  score_kernel<<<NROWS / 256, 256, 0, stream>>>(Hn, Wa1, ba1, Wa2, ba2, sArr);
  final_kernel<<<1, 256, 0, stream>>>(Hn, sArr, Wc, bc, (float*)d_out);
}
